// AttentionHead_6786048328376
// MI455X (gfx1250) — matrix operand order repaired and checked by reading, not compiled, not run on hardware
//
#include <hip/hip_runtime.h>
#include <math.h>

#define B_    8
#define C_    512
#define N_    1024
#define HEADS 8
#define DH_   32
#define CH_   64
#define QK_   256

typedef __attribute__((ext_vector_type(16))) __bf16 v16bf;
typedef __attribute__((ext_vector_type(8)))  __bf16 v8bf;
typedef __attribute__((ext_vector_type(8)))  float  v8f;

// ---------------------------------------------------------------------------
// WMMA fragment loaders (wave32, 16-bit operands), contiguous k order shared by A and B.
// A (16x32, MxK): lane L holds row M=L&15; elems 0..15 <-> K=(L<16?0:16)+e.
// B (32x16, KxN): lane L holds col N=L&15; elems 0..15 <-> K=(L<16?0:16)+e.
// ---------------------------------------------------------------------------
static __device__ __forceinline__ v16bf load_a_frag(const __bf16* base, int ld) {
  int lane = threadIdx.x & 31;
  int row  = lane & 15;
  int half = lane >> 4;
  const __bf16* p = base + row * ld;
  v8bf c0 = *(const v8bf*)(p + (half ? 16 : 0));
  v8bf c1 = *(const v8bf*)(p + (half ? 24 : 8));
  return __builtin_shufflevector(c0, c1, 0,1,2,3,4,5,6,7,8,9,10,11,12,13,14,15);
}

static __device__ __forceinline__ v16bf load_b_frag(const __bf16* base, int ld) {
  int lane = threadIdx.x & 31;
  int col  = lane & 15;
  int half = lane >> 4;
  return *(const v16bf*)(base + col * ld + half * 16);
}

static __device__ __forceinline__ v8f wmma_bf16(v16bf a, v16bf b, v8f c) {
  return __builtin_amdgcn_wmma_f32_16x16x32_bf16(false, a, false, b, (short)0, c,
                                                 false, false);
}

// ---------------------------------------------------------------------------
// Elementwise f32 -> bf16 (weights)
// ---------------------------------------------------------------------------
__global__ void k_f32_to_bf16(const float* __restrict__ s, __bf16* __restrict__ d, int n) {
  int i = blockIdx.x * blockDim.x + threadIdx.x;
  if (i < n) d[i] = (__bf16)s[i];
}

// ---------------------------------------------------------------------------
// LDS-tiled transpose+convert: x [B,C,N] f32 -> xT [B,N,C] bf16
// ---------------------------------------------------------------------------
__global__ void k_transpose_x(const float* __restrict__ x, __bf16* __restrict__ xT) {
  __shared__ float tile[32][33];
  int b  = blockIdx.z;
  int n0 = blockIdx.x * 32;
  int c0 = blockIdx.y * 32;
  int tx = threadIdx.x, ty = threadIdx.y;
  const float* src = x + ((size_t)b * C_ + c0) * N_ + n0;
#pragma unroll
  for (int j = 0; j < 32; j += 8)
    tile[ty + j][tx] = src[(ty + j) * N_ + tx];
  __syncthreads();
  __bf16* dst = xT + ((size_t)b * N_ + n0) * C_ + c0;
#pragma unroll
  for (int j = 0; j < 32; j += 8)
    dst[(ty + j) * C_ + tx] = (__bf16)tile[tx][ty + j];
}

// ---------------------------------------------------------------------------
// Register-blocked GEMM core: one wave computes a 32(M) x 64(N) output tile.
// Per K-step: 2 A-frags + 4 B-frags feed 8 WMMAs (2.7x less fragment traffic
// per WMMA than the 16x16-per-wave scheme -> relieves L2 bandwidth).
// ---------------------------------------------------------------------------
struct Acc2x4 { v8f t[2][4]; };

static __device__ __forceinline__ Acc2x4 gemm_32x64(const __bf16* __restrict__ wbase,
                                                    const __bf16* __restrict__ xbase) {
  Acc2x4 A;
#pragma unroll
  for (int i = 0; i < 2; ++i)
#pragma unroll
    for (int j = 0; j < 4; ++j) A.t[i][j] = (v8f){};
  for (int c = 0; c < C_; c += 32) {
    __builtin_prefetch(xbase + c + 128, 0, 3);  // near-cache prefetch
    v16bf a0 = load_a_frag(wbase + c, C_);
    v16bf a1 = load_a_frag(wbase + 16 * C_ + c, C_);
    v16bf b0 = load_b_frag(xbase + 0 * 16 * C_ + c, C_);
    v16bf b1 = load_b_frag(xbase + 1 * 16 * C_ + c, C_);
    v16bf b2 = load_b_frag(xbase + 2 * 16 * C_ + c, C_);
    v16bf b3 = load_b_frag(xbase + 3 * 16 * C_ + c, C_);
    A.t[0][0] = wmma_bf16(a0, b0, A.t[0][0]);
    A.t[0][1] = wmma_bf16(a0, b1, A.t[0][1]);
    A.t[0][2] = wmma_bf16(a0, b2, A.t[0][2]);
    A.t[0][3] = wmma_bf16(a0, b3, A.t[0][3]);
    A.t[1][0] = wmma_bf16(a1, b0, A.t[1][0]);
    A.t[1][1] = wmma_bf16(a1, b1, A.t[1][1]);
    A.t[1][2] = wmma_bf16(a1, b2, A.t[1][2]);
    A.t[1][3] = wmma_bf16(a1, b3, A.t[1][3]);
  }
  return A;
}

// ---------------------------------------------------------------------------
// Q/K projections: W[256,512] x xT, epilogue += bias + PE, store transposed
// q/k layout [B, HEADS, N, DH] bf16. 32x64 output tile per wave.
// ---------------------------------------------------------------------------
__global__ void k_proj_qk(const __bf16* __restrict__ wq, const __bf16* __restrict__ wk,
                          const __bf16* __restrict__ xT,
                          const float* __restrict__ bq, const float* __restrict__ bk,
                          const float* __restrict__ PE,
                          __bf16* __restrict__ qout, __bf16* __restrict__ kout) {
  int wave = threadIdx.x >> 5, lane = threadIdx.x & 31;
  int id    = blockIdx.x * 8 + wave;  // 0..2047 (q: 0..1023, k: 1024..2047)
  int which = id >> 10;
  int t     = id & 1023;
  int b     = t >> 7;
  int rem   = t & 127;
  int oBase = (rem >> 4) << 5;   // 0..224, step 32
  int nBase = (rem & 15) << 6;   // 0..960, step 64
  const __bf16* W    = which ? wk : wq;
  const float*  bias = which ? bk : bq;
  __bf16*       out  = which ? kout : qout;

  Acc2x4 A = gemm_32x64(W + oBase * C_, xT + (b * N_ + nBase) * C_);

  int col = lane & 15, half = lane >> 4;
#pragma unroll
  for (int i = 0; i < 2; ++i)
#pragma unroll
    for (int j = 0; j < 4; ++j) {
      int nIdx = nBase + j * 16 + col;
#pragma unroll
      for (int r = 0; r < 8; ++r) {
        int o = oBase + i * 16 + r + 8 * half;
        float v = A.t[i][j][r] + bias[o] + PE[(o & 31) * N_ + nIdx];
        int h = o >> 5, d = o & 31;
        out[((b * HEADS + h) * N_ + nIdx) * DH_ + d] = (__bf16)v;
      }
    }
}

// ---------------------------------------------------------------------------
// V projection: Wv[512,512] x xT, += bv, store NATURAL layout v [B, C, N] bf16
// ---------------------------------------------------------------------------
__global__ void k_proj_v(const __bf16* __restrict__ wv, const __bf16* __restrict__ xT,
                         const float* __restrict__ bv, __bf16* __restrict__ vout) {
  int wave = threadIdx.x >> 5, lane = threadIdx.x & 31;
  int id    = blockIdx.x * 8 + wave;  // 0..2047
  int b     = id >> 8;
  int rem   = id & 255;
  int oBase = (rem >> 4) << 5;
  int nBase = (rem & 15) << 6;

  Acc2x4 A = gemm_32x64(wv + oBase * C_, xT + (b * N_ + nBase) * C_);

  int col = lane & 15, half = lane >> 4;
#pragma unroll
  for (int i = 0; i < 2; ++i)
#pragma unroll
    for (int j = 0; j < 4; ++j) {
      int nIdx = nBase + j * 16 + col;
#pragma unroll
      for (int r = 0; r < 8; ++r) {
        int o = oBase + i * 16 + r + 8 * half;
        vout[(b * C_ + o) * N_ + nIdx] = (__bf16)(A.t[i][j][r] + bv[o]);
      }
    }
}

// ---------------------------------------------------------------------------
// Pass 1: per-column (query-axis) softmax stats. One wave owns 16 columns m,
// K-fragment is loop-invariant; sweep all 1024 query rows, one WMMA each,
// online (max, sum-exp) with lane<->lane+16 pair reduction.
// ---------------------------------------------------------------------------
__global__ void k_stats(const __bf16* __restrict__ qb, const __bf16* __restrict__ kb,
                        float* __restrict__ colmax, float* __restrict__ colsum) {
  int wave = threadIdx.x >> 5, lane = threadIdx.x & 31;
  int id    = blockIdx.x * 8 + wave;  // 0..4095
  int b     = id >> 9;
  int rem   = id & 511;
  int h     = rem >> 6;
  int mBase = (rem & 63) << 4;
  int bh    = b * HEADS + h;
  const float scale = 0.17677669529663687f;  // 1/sqrt(32)

  v16bf kf = load_b_frag(kb + (bh * N_ + mBase) * DH_, DH_);
  float runmax = -3.0e38f, runsum = 0.0f;
  for (int n = 0; n < N_; n += 16) {
    v16bf qa = load_a_frag(qb + (bh * N_ + n) * DH_, DH_);
    v8f z = {};
    v8f s = wmma_bf16(qa, kf, z);
    float tmax = s[0] * scale;
#pragma unroll
    for (int r = 1; r < 8; ++r) tmax = fmaxf(tmax, s[r] * scale);
    tmax = fmaxf(tmax, __shfl_xor(tmax, 16, 32));
    float nm = fmaxf(runmax, tmax);
    float tsum = 0.0f;
#pragma unroll
    for (int r = 0; r < 8; ++r) tsum += __expf(s[r] * scale - nm);
    tsum += __shfl_xor(tsum, 16, 32);
    runsum = runsum * __expf(runmax - nm) + tsum;
    runmax = nm;
  }
  if (lane < 16) {
    colmax[bh * N_ + mBase + lane] = runmax;
    colsum[bh * N_ + mBase + lane] = runsum;
  }
}

// ---------------------------------------------------------------------------
// Pass 2: recompute scores, normalize with stats, LDS round-trip to re-layout
// P as A-fragments, then P(32x32) x V(32x64) per wave, accumulated over m.
// Each wave owns 32 query rows: every k-frag feeds 2 score WMMAs and every
// v-frag feeds 2 P*V WMMAs. Store attnT [B, N, C] bf16.
// ---------------------------------------------------------------------------
__global__ void k_attn_out(const __bf16* __restrict__ qb, const __bf16* __restrict__ kb,
                           const __bf16* __restrict__ vb,
                           const float* __restrict__ colmax, const float* __restrict__ colsum,
                           __bf16* __restrict__ attnT) {
  __shared__ __bf16 pbuf[8][32][40];  // per-wave 32(n) x 32(m) bf16, padded (20 KB)
  int wave = threadIdx.x >> 5, lane = threadIdx.x & 31;
  int blk  = blockIdx.x;              // 0..255
  int b    = blk >> 5;
  int rem  = blk & 31;
  int h    = rem >> 2;
  int nBlk = rem & 3;
  int bh   = b * HEADS + h;
  int nW   = nBlk * 256 + wave * 32;  // 32 query rows per wave
  int col = lane & 15, half = lane >> 4;
  const float scale = 0.17677669529663687f;

  v16bf qa0 = load_a_frag(qb + (bh * N_ + nW) * DH_, DH_);       // loop-invariant
  v16bf qa1 = load_a_frag(qb + (bh * N_ + nW + 16) * DH_, DH_);
  v8f acc[2][4];
#pragma unroll
  for (int i = 0; i < 2; ++i)
#pragma unroll
    for (int j = 0; j < 4; ++j) acc[i][j] = (v8f){};

  for (int m = 0; m < N_; m += 32) {
#pragma unroll
    for (int sub = 0; sub < 2; ++sub) {
      int mB = m + sub * 16;
      v16bf kf = load_b_frag(kb + (bh * N_ + mB) * DH_, DH_);
      v8f z = {};
      v8f s0 = wmma_bf16(qa0, kf, z);
      v8f s1 = wmma_bf16(qa1, kf, z);
      int mcol = mB + col;
      float cm  = colmax[bh * N_ + mcol];
      float inv = 1.0f / colsum[bh * N_ + mcol];
#pragma unroll
      for (int r = 0; r < 8; ++r) {
        float p0 = __expf(s0[r] * scale - cm) * inv;
        float p1 = __expf(s1[r] * scale - cm) * inv;
        pbuf[wave][r + 8 * half][sub * 16 + col]      = (__bf16)p0;
        pbuf[wave][16 + r + 8 * half][sub * 16 + col] = (__bf16)p1;
      }
    }
    __syncthreads();
    v16bf pa0 = load_a_frag(&pbuf[wave][0][0], 40);
    v16bf pa1 = load_a_frag(&pbuf[wave][16][0], 40);
#pragma unroll
    for (int ct = 0; ct < 4; ++ct) {
      v16bf vf = load_b_frag(vb + (size_t)(b * C_ + h * CH_ + ct * 16) * N_ + m, N_);
      acc[0][ct] = wmma_bf16(pa0, vf, acc[0][ct]);
      acc[1][ct] = wmma_bf16(pa1, vf, acc[1][ct]);
    }
    __syncthreads();
  }
#pragma unroll
  for (int i = 0; i < 2; ++i)
#pragma unroll
    for (int ct = 0; ct < 4; ++ct)
#pragma unroll
      for (int r = 0; r < 8; ++r) {
        int n  = nW + i * 16 + r + 8 * half;
        int ch = h * CH_ + ct * 16 + col;
        attnT[(b * N_ + n) * C_ + ch] = (__bf16)acc[i][ct][r];
      }
}

// ---------------------------------------------------------------------------
// FFN1: W1 x attnT + b1, mish, store z [B, N, C] bf16 (transposed epilogue)
// ---------------------------------------------------------------------------
__global__ void k_ffn1(const __bf16* __restrict__ w1, const __bf16* __restrict__ attnT,
                       const float* __restrict__ b1, __bf16* __restrict__ z) {
  int wave = threadIdx.x >> 5, lane = threadIdx.x & 31;
  int id    = blockIdx.x * 8 + wave;  // 0..2047
  int b     = id >> 8;
  int rem   = id & 255;
  int oBase = (rem >> 4) << 5;
  int nBase = (rem & 15) << 6;

  Acc2x4 A = gemm_32x64(w1 + oBase * C_, attnT + (b * N_ + nBase) * C_);

  int col = lane & 15, half = lane >> 4;
#pragma unroll
  for (int i = 0; i < 2; ++i)
#pragma unroll
    for (int j = 0; j < 4; ++j) {
      int nIdx = nBase + j * 16 + col;
#pragma unroll
      for (int r = 0; r < 8; ++r) {
        int o = oBase + i * 16 + r + 8 * half;
        float t  = A.t[i][j][r] + b1[o];
        float sp = log1pf(__expf(t));     // softplus
        float mv = t * tanhf(sp);         // mish
        z[(b * N_ + nIdx) * C_ + o] = (__bf16)mv;
      }
    }
}

// ---------------------------------------------------------------------------
// FFN2: W2 x z + b2 + residual(x), f32 output [B, C, N] (natural layout)
// ---------------------------------------------------------------------------
__global__ void k_ffn2(const __bf16* __restrict__ w2, const __bf16* __restrict__ zin,
                       const float* __restrict__ b2, const float* __restrict__ x,
                       float* __restrict__ out) {
  int wave = threadIdx.x >> 5, lane = threadIdx.x & 31;
  int id    = blockIdx.x * 8 + wave;  // 0..2047
  int b     = id >> 8;
  int rem   = id & 255;
  int oBase = (rem >> 4) << 5;
  int nBase = (rem & 15) << 6;

  Acc2x4 A = gemm_32x64(w2 + oBase * C_, zin + (b * N_ + nBase) * C_);

  int col = lane & 15, half = lane >> 4;
#pragma unroll
  for (int i = 0; i < 2; ++i)
#pragma unroll
    for (int j = 0; j < 4; ++j) {
      int nIdx = nBase + j * 16 + col;
#pragma unroll
      for (int r = 0; r < 8; ++r) {
        int o = oBase + i * 16 + r + 8 * half;
        size_t idx = (size_t)(b * C_ + o) * N_ + nIdx;
        out[idx] = A.t[i][j][r] + b2[o] + x[idx];
      }
    }
}

// ---------------------------------------------------------------------------
extern "C" void kernel_launch(void* const* d_in, const int* in_sizes, int n_in,
                              void* d_out, int out_size, void* d_ws, size_t ws_size,
                              hipStream_t stream) {
  (void)in_sizes; (void)n_in; (void)out_size; (void)ws_size;
  const float* x  = (const float*)d_in[0];
  const float* PE = (const float*)d_in[1];
  const float* Wq = (const float*)d_in[2];
  const float* bq = (const float*)d_in[3];
  const float* Wk = (const float*)d_in[4];
  const float* bk = (const float*)d_in[5];
  const float* Wv = (const float*)d_in[6];
  const float* bv = (const float*)d_in[7];
  const float* W1 = (const float*)d_in[8];
  const float* b1 = (const float*)d_in[9];
  const float* W2 = (const float*)d_in[10];
  const float* b2 = (const float*)d_in[11];
  float* out = (float*)d_out;

  char* w = (char*)d_ws;
  size_t off = 0;
  auto alloc = [&](size_t bytes) -> void* {
    void* p = w + off;
    off = (off + bytes + 255) & ~(size_t)255;
    return p;
  };
  __bf16* xT    = (__bf16*)alloc((size_t)B_ * N_ * C_ * 2);
  __bf16* wq_bf = (__bf16*)alloc((size_t)QK_ * C_ * 2);
  __bf16* wk_bf = (__bf16*)alloc((size_t)QK_ * C_ * 2);
  __bf16* wv_bf = (__bf16*)alloc((size_t)C_ * C_ * 2);
  __bf16* w1_bf = (__bf16*)alloc((size_t)C_ * C_ * 2);
  __bf16* w2_bf = (__bf16*)alloc((size_t)C_ * C_ * 2);
  __bf16* q_bf  = (__bf16*)alloc((size_t)B_ * HEADS * N_ * DH_ * 2);
  __bf16* k_bf  = (__bf16*)alloc((size_t)B_ * HEADS * N_ * DH_ * 2);
  __bf16* v_bf  = (__bf16*)alloc((size_t)B_ * C_ * N_ * 2);
  float*  cmax  = (float*)alloc((size_t)B_ * HEADS * N_ * 4);
  float*  csum  = (float*)alloc((size_t)B_ * HEADS * N_ * 4);
  __bf16* attnT = (__bf16*)alloc((size_t)B_ * N_ * C_ * 2);
  __bf16* z_bf  = (__bf16*)alloc((size_t)B_ * N_ * C_ * 2);

  // Stage 0: weight conversion + x transpose/convert
  k_f32_to_bf16<<<(QK_ * C_ + 255) / 256, 256, 0, stream>>>(Wq, wq_bf, QK_ * C_);
  k_f32_to_bf16<<<(QK_ * C_ + 255) / 256, 256, 0, stream>>>(Wk, wk_bf, QK_ * C_);
  k_f32_to_bf16<<<(C_ * C_ + 255) / 256, 256, 0, stream>>>(Wv, wv_bf, C_ * C_);
  k_f32_to_bf16<<<(C_ * C_ + 255) / 256, 256, 0, stream>>>(W1, w1_bf, C_ * C_);
  k_f32_to_bf16<<<(C_ * C_ + 255) / 256, 256, 0, stream>>>(W2, w2_bf, C_ * C_);
  k_transpose_x<<<dim3(N_ / 32, C_ / 32, B_), dim3(32, 8), 0, stream>>>(x, xT);

  // Stage 1: projections (register-blocked WMMA GEMMs)
  k_proj_qk<<<256, 256, 0, stream>>>(wq_bf, wk_bf, xT, bq, bk, PE, q_bf, k_bf);
  k_proj_v<<<256, 256, 0, stream>>>(wv_bf, xT, bv, v_bf);

  // Stage 2: query-axis softmax stats, then recompute + P*V
  k_stats<<<512, 256, 0, stream>>>(q_bf, k_bf, cmax, csum);
  k_attn_out<<<256, 256, 0, stream>>>(q_bf, k_bf, v_bf, cmax, csum, attnT);

  // Stage 3: FFN convs with fused mish / bias+residual
  k_ffn1<<<256, 256, 0, stream>>>(w1_bf, attnT, b1, z_bf);
  k_ffn2<<<256, 256, 0, stream>>>(w2_bf, z_bf, b2, x, out);
}
